// LSTMLayer_37881611551088
// MI455X (gfx1250) — matrix operand order repaired and checked by reading, not compiled, not run on hardware
//
#include <hip/hip_runtime.h>
#include <hip/hip_bf16.h>
#include <math.h>

// ---------------------------------------------------------------------------
// LSTM layer: B=64, T=512, D=1024, U=1024.
// g = [x_t ; h_{t-1}] @ [W ; Umat] + b   (M=64, N=4096, K=2048, bf16 WMMA)
// h_t = sig(f)*h + sig(i)*tanh(c);  y_t = sig(o)*tanh(h_t)
// Weights are pre-packed once into WMMA B-fragment layout (bf16);
// h is kept fp32 (accuracy) + bf16 A-fragment copy (for WMMA).
// ---------------------------------------------------------------------------

typedef __attribute__((ext_vector_type(16))) __bf16 v16bf;
typedef __attribute__((ext_vector_type(8)))  float  v8f;

#define B_  64
#define T_  512
#define D_  1024
#define U_  1024
#define G4  4096   // 4*U (gate concat)
#define K2  2048   // D + U

// workspace layout (bytes)
#define WPACK_OFF 0u          // bf16, 2048*4096*2 = 16 MB  (B fragments)
#define G_OFF     16777216u   // f32,  64*4096*4   = 1 MB   (gate pre-activations)
#define HF_OFF    17825792u   // f32,  64*1024*4   = 256 KB (h state, fp32)
#define HB_OFF    18087936u   // bf16, 64*1024*2   = 128 KB (h state, A fragments)

static __device__ __forceinline__ float sigf(float x) {
    return 1.0f / (1.0f + __expf(-x));
}

// ---------------------------------------------------------------------------
// Pack [W ; Umat] (K=2048 rows, N=4096 cols) into bf16 WMMA B-fragment layout.
// Fragment (kt, nt) = 32x16 tile; stored at frag index (nt*64 + kt), 512 elems.
// Within fragment: lane = (N&15) + (K&8 ? 16 : 0); j = (K&7) + (K&16 ? 8 : 0)  (16 bf16/lane).
// One thread per source element -> coalesced fp32 reads along N.
// ---------------------------------------------------------------------------
__global__ void __launch_bounds__(256)
pack_weights(const float* __restrict__ fw, const float* __restrict__ iw,
             const float* __restrict__ cw, const float* __restrict__ ow,
             const float* __restrict__ fu, const float* __restrict__ iu,
             const float* __restrict__ cu, const float* __restrict__ ou,
             __bf16* __restrict__ wpack)
{
    unsigned tid = blockIdx.x * blockDim.x + threadIdx.x;   // 2048*4096 threads
    unsigned K = tid >> 12;          // 0..2047
    unsigned N = tid & 4095u;        // 0..4095
    unsigned gate = N >> 10;         // f,i,c,o
    unsigned col  = N & 1023u;
    const float* src;
    if (K < 1024u) {   // input-to-gate weights (D rows)
        src = (gate == 0) ? fw : (gate == 1) ? iw : (gate == 2) ? cw : ow;
        src += (size_t)K * 1024u + col;
    } else {           // recurrent weights (U rows)
        src = (gate == 0) ? fu : (gate == 1) ? iu : (gate == 2) ? cu : ou;
        src += (size_t)(K - 1024u) * 1024u + col;
    }
    float v = *src;
    unsigned nt = N >> 4, nl = N & 15u;
    unsigned kt = K >> 5, kr = K & 31u;
    unsigned lane = nl + ((kr & 8u) ? 16u : 0u);
    unsigned j    = (kr & 7u) + ((kr & 16u) ? 8u : 0u);
    wpack[(size_t)(nt * 64u + kt) * 512u + lane * 16u + j] = (__bf16)v;
}

// ---------------------------------------------------------------------------
// h state init: broadcast h0 over batch; fp32 copy + bf16 A-fragment copy.
// A-fragment (16x32 bf16) layout: lanes 0-15 hold K{0..7,16..23} of row M=lane,
// lanes 16-31 hold K{8..15,24..31} of row M=lane-16.
// ---------------------------------------------------------------------------
__global__ void __launch_bounds__(256)
init_h(const float* __restrict__ h0, float* __restrict__ hf,
       __bf16* __restrict__ hbf)
{
    unsigned tid = blockIdx.x * blockDim.x + threadIdx.x;   // 64*1024 threads
    unsigned u = tid & 1023u, b = tid >> 10;
    float v = h0[u];
    hf[tid] = v;
    unsigned mt = b >> 4, m = b & 15u, kt = u >> 5, k = u & 31u;
    unsigned q = k >> 3, r = k & 7u;
    unsigned lane = m + ((q & 1u) ? 16u : 0u);
    unsigned j    = r + ((q & 2u) ? 8u : 0u);
    hbf[(size_t)(mt * 32u + kt) * 512u + lane * 16u + j] = (__bf16)v;
}

// ---------------------------------------------------------------------------
// One recurrence step GEMM: g = [x_t ; h] @ packed weights.
// 1024 waves, one 16x16 f32 output tile per wave, 64 chained bf16 WMMAs.
// ---------------------------------------------------------------------------
__global__ void __launch_bounds__(256)
lstm_gemm_step(const float* __restrict__ x, const __bf16* __restrict__ wpack,
               const __bf16* __restrict__ hbf, float* __restrict__ g, int t)
{
    unsigned lane  = threadIdx.x & 31u;
    unsigned wave  = blockIdx.x * (blockDim.x >> 5) + (threadIdx.x >> 5); // 0..1023
    unsigned mt    = wave & 3u;     // M tile (batch / 16)
    unsigned nt    = wave >> 2;     // N tile (gate col / 16)
    unsigned m     = lane & 15u;
    unsigned khalf = lane >> 4;     // which K-half this lane holds in A frags

    v8f acc = {};
    const __bf16* bptr = wpack + (size_t)(nt * 64u) * 512u + lane * 16u;

    // ---- Part 1: x_t contribution (K tiles 0..31); convert fp32->bf16 here.
    // Lane needs K = kt*32 + khalf*8 + {0..7} and + {16..23} of row b.
    const float* xrow = x + ((size_t)(mt * 16u + m) * T_ + (unsigned)t) * D_
                          + khalf * 8u;
    for (unsigned kt = 0; kt < 32u; ++kt) {
        const float* xp = xrow + kt * 32u;
        float4 p0 = *(const float4*)(xp);
        float4 p1 = *(const float4*)(xp + 4);
        float4 p2 = *(const float4*)(xp + 16);
        float4 p3 = *(const float4*)(xp + 20);
        float fa[16] = { p0.x, p0.y, p0.z, p0.w, p1.x, p1.y, p1.z, p1.w,
                         p2.x, p2.y, p2.z, p2.w, p3.x, p3.y, p3.z, p3.w };
        union { v16bf v; __bf16 e[16]; } a;
        #pragma unroll
        for (int j = 0; j < 16; ++j) a.e[j] = (__bf16)fa[j];
        v16bf bm = *(const v16bf*)(bptr + (size_t)kt * 512u);
        acc = __builtin_amdgcn_wmma_f32_16x16x32_bf16(
                  false, a.v, false, bm, (short)0, acc, false, false);
    }

    // ---- Part 2: h contribution (K tiles 32..63); A fragments pre-packed bf16.
    const __bf16* haf = hbf + (size_t)(mt * 32u) * 512u + lane * 16u;
    for (unsigned kt = 0; kt < 32u; ++kt) {
        v16bf am = *(const v16bf*)(haf + (size_t)kt * 512u);
        v16bf bm = *(const v16bf*)(bptr + (size_t)(32u + kt) * 512u);
        acc = __builtin_amdgcn_wmma_f32_16x16x32_bf16(
                  false, am, false, bm, (short)0, acc, false, false);
    }

    // ---- Store C (16x16 f32): VGPR r -> M = r + (lane>=16 ? 8 : 0), N = lane&15.
    unsigned n  = nt * 16u + (lane & 15u);
    unsigned rb = mt * 16u + ((lane >> 4) ? 8u : 0u);
    #pragma unroll
    for (int r = 0; r < 8; ++r)
        g[(size_t)(rb + r) * G4 + n] = acc[r];
}

// ---------------------------------------------------------------------------
// One recurrence step elementwise: gates -> h update -> y_t.
// ---------------------------------------------------------------------------
__global__ void __launch_bounds__(256)
lstm_ewise_step(const float* __restrict__ g,
                const float* __restrict__ fb, const float* __restrict__ ib,
                const float* __restrict__ cb, const float* __restrict__ ob,
                float* __restrict__ hf, __bf16* __restrict__ hbf,
                float* __restrict__ out, int t)
{
    unsigned tid = blockIdx.x * blockDim.x + threadIdx.x;   // 64*1024 threads
    unsigned u = tid & 1023u, b = tid >> 10;
    size_t gi = (size_t)b * G4 + u;
    float f = g[gi]         + fb[u];
    float i = g[gi + 1024u] + ib[u];
    float c = g[gi + 2048u] + cb[u];
    float o = g[gi + 3072u] + ob[u];
    float hp = hf[tid];
    float hn = sigf(f) * hp + sigf(i) * tanhf(c);
    float y  = sigf(o) * tanhf(hn);
    hf[tid] = hn;
    // bf16 A-fragment copy for next step's WMMA reads
    unsigned mt = b >> 4, m = b & 15u, kt = u >> 5, k = u & 31u;
    unsigned q = k >> 3, r = k & 7u;
    unsigned lane = m + ((q & 1u) ? 16u : 0u);
    unsigned j    = r + ((q & 2u) ? 8u : 0u);
    hbf[(size_t)(mt * 32u + kt) * 512u + lane * 16u + j] = (__bf16)hn;
    out[((size_t)b * T_ + (unsigned)t) * U_ + u] = y;
}

// ---------------------------------------------------------------------------
extern "C" void kernel_launch(void* const* d_in, const int* in_sizes, int n_in,
                              void* d_out, int out_size, void* d_ws, size_t ws_size,
                              hipStream_t stream)
{
    const float* x  = (const float*)d_in[0];
    const float* fw = (const float*)d_in[1];
    const float* fu = (const float*)d_in[2];
    const float* fb = (const float*)d_in[3];
    const float* iw = (const float*)d_in[4];
    const float* iu = (const float*)d_in[5];
    const float* ib = (const float*)d_in[6];
    const float* cw = (const float*)d_in[7];
    const float* cu = (const float*)d_in[8];
    const float* cb = (const float*)d_in[9];
    const float* ow = (const float*)d_in[10];
    const float* ou = (const float*)d_in[11];
    const float* ob = (const float*)d_in[12];
    const float* h0 = (const float*)d_in[13];

    char*   ws    = (char*)d_ws;
    __bf16* wpack = (__bf16*)(ws + WPACK_OFF);
    float*  g     = (float*) (ws + G_OFF);
    float*  hf    = (float*) (ws + HF_OFF);
    __bf16* hbf   = (__bf16*)(ws + HB_OFF);
    float*  out   = (float*)d_out;

    // one-time per launch: pack weights to bf16 B-fragments, init h state
    pack_weights<<<(2048u * 4096u) / 256u, 256, 0, stream>>>(
        fw, iw, cw, ow, fu, iu, cu, ou, wpack);
    init_h<<<(64u * 1024u) / 256u, 256, 0, stream>>>(h0, hf, hbf);

    // sequential recurrence: weights stay L2-resident (16 MB << 192 MB L2)
    for (int t = 0; t < T_; ++t) {
        lstm_gemm_step<<<128, 256, 0, stream>>>(x, wpack, hbf, g, t);
        lstm_ewise_step<<<256, 256, 0, stream>>>(g, fb, ib, cb, ob, hf, hbf, out, t);
    }
}
